// QLSTMGen384_65481071400882
// MI455X (gfx1250) — matrix operand order repaired and checked by reading, not compiled, not run on hardware
//
#include <hip/hip_runtime.h>
#include <hip/hip_bf16.h>

typedef __attribute__((ext_vector_type(16))) _Float16 v16h;
typedef __attribute__((ext_vector_type(8)))  float    v8f;

// Problem sizes (from reference): T=512, B=256, D=128, H=256, K=D+H=384, 4H=1024
constexpr int T_STEPS = 512;
constexpr int BATCH   = 256;
constexpr int DIM     = 128;
constexpr int HID     = 256;
constexpr int KTOT    = 384;   // D + H
constexpr int NPACK   = 1024;  // 4H, gate-interleaved: col = 4*j + gate

// Block tiling for the per-step GEMM
constexpr int BM = 64;    // batch rows per block
constexpr int BN = 128;   // packed gate-columns per block (= 32 h-units * 4 gates)
constexpr int SA_STRIDE = 40;   // halves per row (32 payload + 8 pad) -> conflict-free b128
constexpr int SB_STRIDE = 40;
constexpr int SZ_STRIDE = 132;  // floats per row (128 payload + 4 pad)

__device__ __forceinline__ float sigmoidf_fast(float x) {
  return 1.0f / (1.0f + __expf(-x));
}
__device__ __forceinline__ float tanhf_fast(float x) {
  float e2 = __expf(2.0f * x);
  return (e2 - 1.0f) / (e2 + 1.0f);
}

// ---------------------------------------------------------------------------
// Prep: pack W (gate-interleaved) to f16, pack biases
// ---------------------------------------------------------------------------
__global__ __launch_bounds__(256) void pack_w_kernel(
    const float* __restrict__ Wf, const float* __restrict__ Wi,
    const float* __restrict__ Wg, const float* __restrict__ Wo,
    const float* __restrict__ bf, const float* __restrict__ bi,
    const float* __restrict__ bg, const float* __restrict__ bo,
    _Float16* __restrict__ Wp, float* __restrict__ bp) {
  int idx = blockIdx.x * 256 + threadIdx.x;          // [0, 1024*384)
  int col = idx / KTOT;
  int k   = idx - col * KTOT;
  int j   = col >> 2;
  int g   = col & 3;
  const float* W = (g == 0) ? Wf : (g == 1) ? Wi : (g == 2) ? Wg : Wo;
  Wp[col * KTOT + k] = (_Float16)W[j * KTOT + k];
  if (idx < NPACK) {
    int jj = idx >> 2, gg = idx & 3;
    const float* bb = (gg == 0) ? bf : (gg == 1) ? bi : (gg == 2) ? bg : bo;
    bp[idx] = bb[jj];
  }
}

__global__ __launch_bounds__(256) void init_state_kernel(
    _Float16* __restrict__ h16, float* __restrict__ c32) {
  int i = blockIdx.x * 256 + threadIdx.x;            // [0, B*H)
  h16[i] = (_Float16)0.0f;
  c32[i] = 0.0f;
}

// ---------------------------------------------------------------------------
// One LSTM time step: z = [x_t, h] @ Wp^T + b  then gate update (fused)
// grid = (4, 8): x -> batch tile (64 rows), y -> packed column tile (128 cols)
// ---------------------------------------------------------------------------
__global__ __launch_bounds__(256, 1) void lstm_step_kernel(
    const float*    __restrict__ x,     // [B, D] slice for this t
    const _Float16* __restrict__ Wp,    // [1024, 384] gate-interleaved
    const float*    __restrict__ bp,    // [1024]
    const _Float16* __restrict__ hin,   // [B, H] (previous h, f16)
    _Float16*       __restrict__ hout,  // [B, H] (new h, f16)
    float*          __restrict__ h32,   // [B, H] (new h, f32)
    float*          __restrict__ c32,   // [B, H] (cell state, in/out)
    _Float16*       __restrict__ hs)    // [B, H] slice of hs for this t
{
  __shared__ _Float16 sA[BM * SA_STRIDE];
  __shared__ _Float16 sB[BN * SB_STRIDE];
  __shared__ float    sZ[BM * SZ_STRIDE];

  const int tid  = threadIdx.x;
  const int wave = tid >> 5;
  const int lane = tid & 31;
  const int lm   = lane & 15;
  const int hi   = lane >> 4;

  const int m0blk = blockIdx.x * BM;    // batch row base
  const int n0blk = blockIdx.y * BN;    // packed column base

  const int wm = wave & 1;              // wave's 32-row sub-tile
  const int wn = wave >> 1;             // wave's 32-col sub-tile

  v8f acc[2][2] = {};                   // 4x 16x16 f32 accumulators

  for (int kc = 0; kc < KTOT; kc += 32) {
    __syncthreads();  // previous iteration's fragment reads done

    // ---- stage A chunk: combined[64 rows][32 k] into LDS (f16) ----
    {
      int r = tid >> 2, seg = tid & 3;          // 8 halves per thread
      int b = m0blk + r;
      int k = kc + seg * 8;
      _Float16* dst = &sA[r * SA_STRIDE + seg * 8];
      if (kc < DIM) {
        const float* src = x + b * DIM + k;     // x part, convert f32->f16
        _Float16 tmp[8];
#pragma unroll
        for (int i = 0; i < 8; ++i) tmp[i] = (_Float16)src[i];
        *(float4*)dst = *(const float4*)tmp;
      } else {
        const _Float16* src = hin + b * HID + (k - DIM);
        *(float4*)dst = *(const float4*)src;    // 16B f16 copy
      }
    }
    // ---- stage B chunk: Wp[128 cols][32 k] into LDS ----
    {
      int c = tid >> 1, seg = tid & 1;          // 16 halves per thread
      const _Float16* src = Wp + (n0blk + c) * KTOT + kc + seg * 16;
      _Float16* dst = &sB[c * SB_STRIDE + seg * 16];
      *(float4*)(dst)     = *(const float4*)(src);
      *(float4*)(dst + 8) = *(const float4*)(src + 8);
    }
    __syncthreads();

    // ---- build fragments per ISA 16-bit A(16x32)/B(32x16) layouts ----
    union AF { v16h v; float4 q[2]; } afrag[2];
#pragma unroll
    for (int i = 0; i < 2; ++i) {
      const _Float16* base = &sA[(wm * 32 + i * 16 + lm) * SA_STRIDE];
      afrag[i].q[0] = *(const float4*)(base + hi * 8);        // K 0..7 / 8..15
      afrag[i].q[1] = *(const float4*)(base + 16 + hi * 8);   // K 16..23 / 24..31
    }
    union BF { v16h v; float4 q[2]; } bfrag[2];
#pragma unroll
    for (int j = 0; j < 2; ++j) {
      const _Float16* base = &sB[(wn * 32 + j * 16 + lm) * SB_STRIDE + hi * 8];
      bfrag[j].q[0] = *(const float4*)(base);                 // K 0..7 / 8..15
      bfrag[j].q[1] = *(const float4*)(base + 16);            // K 16..23 / 24..31
    }

#pragma unroll
    for (int i = 0; i < 2; ++i)
#pragma unroll
      for (int j = 0; j < 2; ++j)
        acc[i][j] = __builtin_amdgcn_wmma_f32_16x16x32_f16(
            false, afrag[i].v, false, bfrag[j].v,
            (short)0, acc[i][j], false, false);
  }

  // ---- spill z tile to LDS (padded stride -> conflict-free) ----
#pragma unroll
  for (int i = 0; i < 2; ++i)
#pragma unroll
    for (int j = 0; j < 2; ++j)
#pragma unroll
      for (int r = 0; r < 8; ++r) {
        int row = wm * 32 + i * 16 + hi * 8 + r;
        int col = wn * 32 + j * 16 + lm;
        sZ[row * SZ_STRIDE + col] = acc[i][j][r];
      }
  __syncthreads();

  // ---- fused LSTM cell update: 64 rows x 32 h-units, 8 per thread ----
#pragma unroll
  for (int it = 0; it < 8; ++it) {
    int q  = it * 256 + tid;
    int r  = q >> 5;                    // local batch row 0..63
    int jj = q & 31;                    // local h-unit 0..31
    int b  = m0blk + r;
    int j  = (n0blk >> 2) + jj;         // global h-unit
    int cb = jj * 4;

    float zf = sZ[r * SZ_STRIDE + cb + 0] + bp[n0blk + cb + 0];
    float zi = sZ[r * SZ_STRIDE + cb + 1] + bp[n0blk + cb + 1];
    float zg = sZ[r * SZ_STRIDE + cb + 2] + bp[n0blk + cb + 2];
    float zo = sZ[r * SZ_STRIDE + cb + 3] + bp[n0blk + cb + 3];

    int gidx = b * HID + j;
    float f  = sigmoidf_fast(zf);
    float ii = sigmoidf_fast(zi);
    float g  = tanhf_fast(zg);
    float o  = sigmoidf_fast(zo);
    float cn = f * c32[gidx] + ii * g;
    float hn = o * tanhf_fast(cn);

    c32[gidx]  = cn;
    h32[gidx]  = hn;
    hout[gidx] = (_Float16)hn;
    hs[gidx]   = (_Float16)hn;
  }
}

// ---------------------------------------------------------------------------
// Classifier: probs[t,b] = sigmoid(dot(hs[t,b,:], Wc) + bc); out [T,B,2]
// One wave32 per (t,b) row, shuffle reduction.
// ---------------------------------------------------------------------------
__global__ __launch_bounds__(256) void classifier_kernel(
    const _Float16* __restrict__ hs,   // [T*B, H] f16
    const float* __restrict__ Wc,      // [H]
    const float* __restrict__ bc,      // [1]
    float* __restrict__ out)           // [T*B, 2]
{
  int row  = blockIdx.x * 8 + (threadIdx.x >> 5);
  int lane = threadIdx.x & 31;
  const _Float16* hrow = hs + (size_t)row * HID;
  float s = 0.0f;
#pragma unroll
  for (int i = 0; i < 8; ++i) {
    int idx = i * 32 + lane;
    s += (float)hrow[idx] * Wc[idx];
  }
#pragma unroll
  for (int off = 16; off > 0; off >>= 1) s += __shfl_xor(s, off);
  if (lane == 0) {
    float p = sigmoidf_fast(s + bc[0]);   // SHIFT == 0
    out[row * 2 + 0] = p;
    out[row * 2 + 1] = 1.0f - p;
  }
}

__global__ __launch_bounds__(256) void finalize_kernel(
    const float* __restrict__ h32, const float* __restrict__ c32,
    float* __restrict__ out) {
  int i = blockIdx.x * 256 + threadIdx.x;      // [0, B*H)
  out[T_STEPS * BATCH * 2 + i]               = h32[i];   // hx
  out[T_STEPS * BATCH * 2 + BATCH * HID + i] = c32[i];   // cx
}

// ---------------------------------------------------------------------------
extern "C" void kernel_launch(void* const* d_in, const int* in_sizes, int n_in,
                              void* d_out, int out_size, void* d_ws, size_t ws_size,
                              hipStream_t stream) {
  (void)in_sizes; (void)n_in; (void)out_size; (void)ws_size;
  const float* x  = (const float*)d_in[0];
  const float* Wf = (const float*)d_in[1];
  const float* bf = (const float*)d_in[2];
  const float* Wi = (const float*)d_in[3];
  const float* bi = (const float*)d_in[4];
  const float* Wg = (const float*)d_in[5];
  const float* bg = (const float*)d_in[6];
  const float* Wo = (const float*)d_in[7];
  const float* bo = (const float*)d_in[8];
  const float* Wc = (const float*)d_in[9];
  const float* bc = (const float*)d_in[10];

  // Workspace layout (16B-aligned offsets)
  char* ws = (char*)d_ws;
  _Float16* Wp  = (_Float16*)(ws + 0);                    //  786432 B
  float*    bp  = (float*)   (ws + 786432);               //    4096 B
  _Float16* hA  = (_Float16*)(ws + 790528);               //  131072 B
  _Float16* hB  = (_Float16*)(ws + 921600);               //  131072 B
  float*    h32 = (float*)   (ws + 1052672);              //  262144 B
  float*    c32 = (float*)   (ws + 1314816);              //  262144 B
  _Float16* hsb = (_Float16*)(ws + 1576960);              // 67108864 B
  float*    out = (float*)d_out;

  pack_w_kernel<<<(NPACK * KTOT) / 256, 256, 0, stream>>>(
      Wf, Wi, Wg, Wo, bf, bi, bg, bo, Wp, bp);
  init_state_kernel<<<(BATCH * HID) / 256, 256, 0, stream>>>(hA, c32);

  for (int t = 0; t < T_STEPS; ++t) {
    const _Float16* hin  = (t & 1) ? hB : hA;
    _Float16*       hout = (t & 1) ? hA : hB;
    lstm_step_kernel<<<dim3(BATCH / BM, NPACK / BN), 256, 0, stream>>>(
        x + (size_t)t * BATCH * DIM, Wp, bp, hin, hout, h32, c32,
        hsb + (size_t)t * BATCH * HID);
  }

  classifier_kernel<<<(T_STEPS * BATCH) / 8, 256, 0, stream>>>(hsb, Wc, bc, out);
  finalize_kernel<<<(BATCH * HID) / 256, 256, 0, stream>>>(h32, c32, out);
}
